// EncoderSRNN_50706383897347
// MI455X (gfx1250) — compile-verified, matrix operand order checked by reading, not run on hardware
//
#include <hip/hip_runtime.h>
#include <math.h>

// ---------------- problem constants ----------------
#define T_    256
#define B_    128
#define E_    256
#define H_    256
#define SD_   64
#define SS_   64
#define NTOP  128   // SDIM*SDEPTH
#define MB    16    // batch rows per scan workgroup
#define NBLK  8     // B_/MB

typedef unsigned short u16;
typedef __bf16 bf16_t;
typedef u16    v16u16 __attribute__((ext_vector_type(16)));
typedef bf16_t v16bf  __attribute__((ext_vector_type(16)));
typedef float  v8f    __attribute__((ext_vector_type(8)));

// ---------------- output layout (floats) ----------------
// outputs [T,B,H] | hid [B,H] | stack [B,SS,SD] | acts [T,B]
#define OUT_OUTPUTS ((size_t)0)
#define OUT_HID     ((size_t)T_*B_*H_)                  // 8388608
#define OUT_STACK   (OUT_HID + (size_t)B_*H_)           // 8421376
#define OUT_ACTS    (OUT_STACK + (size_t)B_*SS_*SD_)    // 8945664

// ---------------- workspace layout (bytes) ----------------
#define WS_ME    ((size_t)0)                         // f32 [T*B*H] = 33554432 B
#define WS_WEH   ((size_t)33554432)                  // bf16 [256][256]
#define WS_WHH   (WS_WEH  + 131072)                  // bf16 [256][256]
#define WS_WSH   (WS_WHH  + 131072)                  // bf16 [256][128]
#define WS_WHS   (WS_WSH  + 65536)                   // bf16 [64][256]
#define WS_WSU   (WS_WHS  + 32768)                   // bf16 [64][128]
#define WS_HEAD  (WS_WSU  + 16384)                   // bf16 [16][256] (ha rows 0-2, hg row 3)

__device__ __forceinline__ u16 f2bf(float f) {
    union { float f; unsigned u; } c; c.f = f;
    unsigned u = c.u;
    unsigned r = u + 0x7FFFu + ((u >> 16) & 1u);
    return (u16)(r >> 16);
}

__device__ __forceinline__ v8f vzero() {
    v8f z;
#pragma unroll
    for (int i = 0; i < 8; ++i) z[i] = 0.0f;
    return z;
}

// One 16x16 f32 tile, A[16xK] (row-major, lda) x B[NxK] row-major as W (ldb),
// computing A * W^T via v_wmma_f32_16x16x32_bf16.  Bm must already point at
// row n0 of W.  Works for generic (LDS or global) pointers.
template <int K>
__device__ __forceinline__ v8f wmma_k(const u16* A, int lda,
                                      const u16* Bm, int ldb,
                                      v8f acc, int lane) {
    const int r = lane & 15;
    const int h = (lane >> 4) << 4;   // K-half offset: 0 or 16
    const u16* ap = A  + r * lda + h;
    const u16* bp = Bm + r * ldb + h;
#pragma unroll
    for (int k = 0; k < K; k += 32) {
        v16u16 au = *(const v16u16*)(ap + k);
        v16u16 bu = *(const v16u16*)(bp + k);
        acc = __builtin_amdgcn_wmma_f32_16x16x32_bf16(
            false, __builtin_bit_cast(v16bf, au),
            false, __builtin_bit_cast(v16bf, bu),
            (short)0, acc, false, false);
    }
    return acc;
}

// ---- Tensor Data Mover: 2-D bf16 tile, global -> LDS -------------------
// D# per CDNA5 ISA: group0 = {count, lds_addr,
// global_addr, type=2}; group1 = {data_size=1 (2B), tensor_dim0=K,
// tensor_dim1=N, tile_dim0=K, tile_dim1=N, tensor_dim0_stride=K}.
// Tracked with TENSORcnt; completed via s_wait_tensorcnt 0.
#if __has_builtin(__builtin_amdgcn_tensor_load_to_lds)
#define SRNN_HAVE_TDM 1
typedef unsigned int u32x4 __attribute__((ext_vector_type(4)));
typedef int          i32x4 __attribute__((ext_vector_type(4)));
typedef int          i32x8 __attribute__((ext_vector_type(8)));

__device__ __forceinline__ void tdm_load_2d(const void* gptr, void* lptr,
                                            unsigned K, unsigned N) {
    const unsigned long long ga = (unsigned long long)gptr;
    const unsigned la = (unsigned)(unsigned long long)lptr;  // flat[31:0] == LDS offset

    u32x4 g0;
    g0[0] = 1u;                                     // count=1, user descriptor
    g0[1] = la;                                     // lds_addr
    g0[2] = (unsigned)(ga & 0xFFFFFFFFu);           // global_addr[31:0]
    g0[3] = (unsigned)((ga >> 32) & 0x01FFFFFFu)    // global_addr[56:32]
          | (2u << 30);                             // type = 2 ("image")

    i32x8 g1;
    g1[0] = (int)(1u << 16);                        // data_size=1 -> 2 bytes/elem
    g1[1] = (int)((K & 0xFFFFu) << 16);             // tensor_dim0[15:0] @ bits 63:48
    g1[2] = (int)((N & 0xFFFFu) << 16);             // tensor_dim1[15:0] @ bits 95:80
    g1[3] = (int)((K & 0xFFFFu) << 16);             // tile_dim0 @ bits 127:112
    g1[4] = (int)(N & 0xFFFFu);                     // tile_dim1 @ bits 143:128
    g1[5] = (int)K;                                 // tensor_dim0_stride[31:0]
    g1[6] = 0;
    g1[7] = 0;

    i32x4 z4; z4[0] = z4[1] = z4[2] = z4[3] = 0;
#if defined(__clang_major__) && (__clang_major__ >= 23)
    i32x8 z8;
#pragma unroll
    for (int i = 0; i < 8; ++i) z8[i] = 0;
    __builtin_amdgcn_tensor_load_to_lds(g0, g1, z4, z4, z8, 0);
#else
    __builtin_amdgcn_tensor_load_to_lds(g0, g1, z4, z4, 0);
#endif
}
#else
#define SRNN_HAVE_TDM 0
#endif

// ================= kernel 0: f32 -> bf16 weight prep =================
__global__ void srnn_prep(const float* __restrict__ W_eh,
                          const float* __restrict__ W_hh,
                          const float* __restrict__ W_sh,
                          const float* __restrict__ W_hs,
                          const float* __restrict__ W_su,
                          const float* __restrict__ W_ha,
                          const float* __restrict__ W_hg,
                          u16* __restrict__ wWeh, u16* __restrict__ wWhh,
                          u16* __restrict__ wWsh, u16* __restrict__ wWhs,
                          u16* __restrict__ wWsu, u16* __restrict__ wHead) {
    int i = blockIdx.x * blockDim.x + threadIdx.x;
    if (i < 65536) { wWeh[i] = f2bf(W_eh[i]); return; }
    i -= 65536;
    if (i < 65536) { wWhh[i] = f2bf(W_hh[i]); return; }
    i -= 65536;
    if (i < 32768) { wWsh[i] = f2bf(W_sh[i]); return; }
    i -= 32768;
    if (i < 16384) { wWhs[i] = f2bf(W_hs[i]); return; }
    i -= 16384;
    if (i < 8192)  { wWsu[i] = f2bf(W_su[i]); return; }
    i -= 8192;
    if (i < 4096) {
        int row = i >> 8, col = i & 255;
        float v = (row < 3) ? W_ha[row * 256 + col]
                : (row == 3) ? W_hg[col] : 0.0f;
        wHead[i] = f2bf(v);
    }
}

// ====== kernel 1: mE[t,b,:] = emb_W[inputs[t,b]] @ W_eh^T + b_eh  ======
// grid (2, T*B/16), block 256 (8 waves); wave w -> N-tile blockIdx.x*8+w
__global__ __launch_bounds__(256, 2)
void srnn_embproj(const int* __restrict__ inputs,
                  const float* __restrict__ emb_W,
                  const u16* __restrict__ wWeh,
                  const float* __restrict__ b_eh,
                  float* __restrict__ mE) {
    __shared__ alignas(32) u16 sA[MB * E_];   // 8 KB gathered bf16 emb tile
    __shared__ int sIdx[MB];

    const int tid  = threadIdx.x;
    const int lane = tid & 31;
    const int wave = tid >> 5;
    const int row0 = blockIdx.y * MB;         // flat (t*B + b) row

    if (tid < MB) sIdx[tid] = inputs[row0 + tid];
    __syncthreads();

    {   // gather + convert: thread -> 16 contiguous f32 of one row
        const int r  = tid >> 4;
        const int c0 = (tid & 15) << 4;
        const size_t base = (size_t)sIdx[r] * E_ + c0;
#pragma unroll
        for (int j = 0; j < 16; ++j)
            sA[r * E_ + c0 + j] = f2bf(emb_W[base + j]);
    }
    __syncthreads();

    const int n0 = blockIdx.x * 128 + wave * 16;
    v8f acc = vzero();
    acc = wmma_k<E_>(sA, E_, wWeh + (size_t)n0 * E_, E_, acc, lane);

    const int n = n0 + (lane & 15);
    const float bb = b_eh[n];
    float* outp = mE + (size_t)row0 * H_ + n;
#pragma unroll
    for (int r = 0; r < 8; ++r) {
        const int m = r + ((lane >> 4) << 3);
        outp[m * H_] = acc[r] + bb;
    }
}

// ================= kernel 2: sequential scan =================
// grid NBLK=8 blocks (16 batch rows each), block 256 threads (8 waves).
// All recurrent weights live in LDS (bf16, staged by the TDM); stack lives
// in global (L2-resident).
__global__ __launch_bounds__(256, 1)
void srnn_scan(const float* __restrict__ mE,
               const u16* __restrict__ wWhh, const u16* __restrict__ wWsh,
               const u16* __restrict__ wWhs, const u16* __restrict__ wWsu,
               const u16* __restrict__ wHead,
               const float* __restrict__ b_hh, const float* __restrict__ b_sh,
               const float* __restrict__ b_hs, const float* __restrict__ b_su,
               const float* __restrict__ b_ha, const float* __restrict__ b_hg,
               const float* __restrict__ empty_elem,
               float* __restrict__ out) {
    __shared__ alignas(32) u16 sWhh[H_ * H_];     // 128 KB
    __shared__ alignas(32) u16 sWsh[H_ * NTOP];   //  64 KB
    __shared__ alignas(32) u16 sWhs[SD_ * H_];    //  32 KB
    __shared__ alignas(32) u16 sWsu[SD_ * NTOP];  //  16 KB
    __shared__ alignas(32) u16 sHead[16 * H_];    //   8 KB
    __shared__ alignas(32) u16 sHid[2][MB * H_];  //  16 KB (double buffer)
    __shared__ alignas(32) u16 sTops[MB * NTOP];  //   4 KB
    __shared__ float sPush[MB * SD_];             //   4 KB
    __shared__ float sU[MB * SD_];                //   4 KB
    __shared__ float sLogit[MB][4];
    __shared__ float sP[MB][4];
    __shared__ float sBmh[H_];
    __shared__ float sBhs[SD_], sBsu[SD_], sBha[4];

    const int tid  = threadIdx.x;
    const int lane = tid & 31;
    const int wave = tid >> 5;
    const int bbase = blockIdx.x * MB;

    // ---- one-time: stage weights into LDS via the Tensor Data Mover ----
#if SRNN_HAVE_TDM
    if (wave == 0) {
        tdm_load_2d(wWhh,  sWhh,  H_,   H_);
        tdm_load_2d(wWsh,  sWsh,  NTOP, H_);
        tdm_load_2d(wWhs,  sWhs,  H_,   SD_);
        tdm_load_2d(wWsu,  sWsu,  NTOP, SD_);
        tdm_load_2d(wHead, sHead, H_,   16);
        __builtin_amdgcn_s_wait_tensorcnt(0);
    }
#else
    {
        const uint4* s; uint4* d; int n4;
        s = (const uint4*)wWhh; d = (uint4*)sWhh; n4 = (H_ * H_) / 8;
        for (int i = tid; i < n4; i += 256) d[i] = s[i];
        s = (const uint4*)wWsh; d = (uint4*)sWsh; n4 = (H_ * NTOP) / 8;
        for (int i = tid; i < n4; i += 256) d[i] = s[i];
        s = (const uint4*)wWhs; d = (uint4*)sWhs; n4 = (SD_ * H_) / 8;
        for (int i = tid; i < n4; i += 256) d[i] = s[i];
        s = (const uint4*)wWsu; d = (uint4*)sWsu; n4 = (SD_ * NTOP) / 8;
        for (int i = tid; i < n4; i += 256) d[i] = s[i];
        s = (const uint4*)wHead; d = (uint4*)sHead; n4 = (16 * H_) / 8;
        for (int i = tid; i < n4; i += 256) d[i] = s[i];
    }
#endif
    // combined biases (b_eh already folded into mE)
    if (tid < H_)  sBmh[tid] = b_hh[tid] + b_sh[tid];
    if (tid < SD_) { sBhs[tid] = b_hs[tid]; sBsu[tid] = b_su[tid]; }
    if (tid < 3)   sBha[tid] = b_ha[tid];
    if (tid == 3)  sBha[3]  = b_hg[0];

    // hid0 = 0
    for (int e = tid; e < MB * H_; e += 256) sHid[0][e] = 0;

    // stack0 = broadcast(empty_elem); tops0 from it
    float* stackG = out + OUT_STACK + (size_t)bbase * SS_ * SD_;
    for (int e = tid; e < MB * SS_ * SD_; e += 256)
        stackG[e] = empty_elem[e & (SD_ - 1)];
    for (int e = tid; e < MB * NTOP; e += 256)
        sTops[e] = f2bf(empty_elem[e & (SD_ - 1)]);
    __syncthreads();

    int cur = 0;
    for (int t = 0; t < T_; ++t) {
        const u16* Ah = sHid[cur];
        u16*       Hn = sHid[cur ^ 1];

        // ---- phase 1: all GEMMs on current hid/tops (WMMA) ----
        // mhid: 16 N-tiles, 2 per wave (W_hh over K=256 + W_sh over K=128)
#pragma unroll
        for (int tt = 0; tt < 2; ++tt) {
            const int n0 = (wave + tt * 8) * 16;
            v8f acc = vzero();
            acc = wmma_k<H_>(Ah, H_, sWhh + n0 * H_, H_, acc, lane);
            acc = wmma_k<NTOP>(sTops, NTOP, sWsh + n0 * NTOP, NTOP, acc, lane);
            const int n = n0 + (lane & 15);
            const float bb = sBmh[n];
            const float* mEp = mE + ((size_t)t * B_ + bbase) * H_ + n;
            float* outp = out + OUT_OUTPUTS + ((size_t)t * B_ + bbase) * H_ + n;
#pragma unroll
            for (int r = 0; r < 8; ++r) {
                const int m = r + ((lane >> 4) << 3);
                float v = acc[r] + bb + mEp[m * H_];
                v = fmaxf(v, 0.0f);
                outp[m * H_] = v;                 // outputs[t]
                Hn[m * H_ + n] = f2bf(v);         // next hid (bf16)
            }
        }
        if (wave < 4) {           // push_val = relu(hid @ W_hs^T + b_hs)
            const int n0 = wave * 16;
            v8f acc = vzero();
            acc = wmma_k<H_>(Ah, H_, sWhs + n0 * H_, H_, acc, lane);
            const int n = n0 + (lane & 15);
            const float bb = sBhs[n];
#pragma unroll
            for (int r = 0; r < 8; ++r)
                sPush[(r + ((lane >> 4) << 3)) * SD_ + n] = fmaxf(acc[r] + bb, 0.0f);
        } else {                  // u_val = relu(tops @ W_su^T + b_su)
            const int n0 = (wave - 4) * 16;
            v8f acc = vzero();
            acc = wmma_k<NTOP>(sTops, NTOP, sWsu + n0 * NTOP, NTOP, acc, lane);
            const int n = n0 + (lane & 15);
            const float bb = sBsu[n];
#pragma unroll
            for (int r = 0; r < 8; ++r)
                sU[(r + ((lane >> 4) << 3)) * SD_ + n] = fmaxf(acc[r] + bb, 0.0f);
        }
        // fused head on the lightest wave: cols 0-2 = act logits, col 3 = gate
        if (wave == 4) {
            v8f acc = vzero();
            acc = wmma_k<H_>(Ah, H_, sHead, H_, acc, lane);
            const int n = lane & 15;
            if (n < 4) {
#pragma unroll
                for (int r = 0; r < 8; ++r)
                    sLogit[r + ((lane >> 4) << 3)][n] = acc[r];
            }
        }
        __syncthreads();

        // ---- phase 2: per-row scalar head (softmax ^ gamma, argmax) ----
        if (tid < MB) {
            const float l0 = sLogit[tid][0] + sBha[0];
            const float l1 = sLogit[tid][1] + sBha[1];
            const float l2 = sLogit[tid][2] + sBha[2];
            const float g  = sLogit[tid][3] + sBha[3];
            const float sp = (g > 20.0f) ? g : log1pf(expf(g));
            const float gamma = 1.0f + sp;
            const float mx = fmaxf(l0, fmaxf(l1, l2));
            const float e0 = expf(l0 - mx), e1 = expf(l1 - mx), e2 = expf(l2 - mx);
            const float inv = 1.0f / (e0 + e1 + e2);
            const float s0 = powf(e0 * inv, gamma);
            const float s1 = powf(e1 * inv, gamma);
            const float s2 = powf(e2 * inv, gamma);
            const float invs = 1.0f / (s0 + s1 + s2 + 1e-16f);
            sP[tid][0] = s0 * invs;
            sP[tid][1] = s1 * invs;
            sP[tid][2] = s2 * invs;
            const int ch = (s0 >= s1 && s0 >= s2) ? 0 : ((s1 >= s2) ? 1 : 2);
            out[OUT_ACTS + (size_t)t * B_ + bbase + tid] = (float)ch;
        }
        __syncthreads();

        // ---- phase 3: soft stack blend (4 columns/thread, rolling window) ----
        {
#pragma unroll
            for (int cc = 0; cc < 4; ++cc) {
                const int c = tid * 4 + cc;            // 0..1023
                const int b = c >> 6, j = c & 63;
                const float pp = sP[b][0], pq = sP[b][1], pn = sP[b][2];
                float* sc = stackG + (size_t)b * SS_ * SD_ + j;
                float prev = sc[0];
                float curr = sc[SD_];
                sc[0] = pp * sPush[b * SD_ + j] + pq * sU[b * SD_ + j] + pn * prev;
                for (int i = 1; i < SS_; ++i) {
                    const float next = (i + 1 < SS_) ? sc[(i + 1) * SD_] : 0.0f;
                    sc[i * SD_] = pp * prev + pq * next + pn * curr;
                    prev = curr; curr = next;
                }
            }
        }
        __threadfence();
        __syncthreads();

        // ---- phase 4: rebuild bf16 tops from stack rows 0..1 ----
        for (int e = tid; e < MB * NTOP; e += 256) {
            const int b = e >> 7, d = e & 127;
            sTops[b * NTOP + d] = f2bf(stackG[(size_t)b * SS_ * SD_ + d]);
        }
        // prefetch next step's mE slab (global_prefetch_b8 path)
        if (t + 1 < T_)
            __builtin_prefetch(mE + ((size_t)(t + 1) * B_ + bbase) * H_ + tid * 16, 0, 1);

        cur ^= 1;
        __syncthreads();
    }

    // final hid = outputs[T-1] (fp32 copy)
    for (int e = tid; e < MB * H_; e += 256) {
        const int m = e >> 8, n = e & 255;
        out[OUT_HID + (size_t)(bbase + m) * H_ + n] =
            out[OUT_OUTPUTS + ((size_t)(T_ - 1) * B_ + bbase + m) * H_ + n];
    }
}

// ================= host launcher =================
extern "C" void kernel_launch(void* const* d_in, const int* in_sizes, int n_in,
                              void* d_out, int out_size, void* d_ws, size_t ws_size,
                              hipStream_t stream) {
    (void)in_sizes; (void)n_in; (void)out_size; (void)ws_size;

    const int*   inputs     = (const int*)  d_in[0];
    const float* emb_W      = (const float*)d_in[1];
    const float* W_hh       = (const float*)d_in[2];
    const float* b_hh       = (const float*)d_in[3];
    const float* W_eh       = (const float*)d_in[4];
    const float* b_eh       = (const float*)d_in[5];
    const float* W_ha       = (const float*)d_in[6];
    const float* b_ha       = (const float*)d_in[7];
    const float* W_hg       = (const float*)d_in[8];
    const float* b_hg       = (const float*)d_in[9];
    const float* W_hs       = (const float*)d_in[10];
    const float* b_hs       = (const float*)d_in[11];
    const float* W_sh       = (const float*)d_in[12];
    const float* b_sh       = (const float*)d_in[13];
    const float* W_su       = (const float*)d_in[14];
    const float* b_su       = (const float*)d_in[15];
    const float* empty_elem = (const float*)d_in[16];

    float* out = (float*)d_out;
    char*  ws  = (char*)d_ws;

    float* mE    = (float*)(ws + WS_ME);
    u16*   wWeh  = (u16*)(ws + WS_WEH);
    u16*   wWhh  = (u16*)(ws + WS_WHH);
    u16*   wWsh  = (u16*)(ws + WS_WSH);
    u16*   wWhs  = (u16*)(ws + WS_WHS);
    u16*   wWsu  = (u16*)(ws + WS_WSU);
    u16*   wHead = (u16*)(ws + WS_HEAD);

    // 0) weight conversion to bf16
    const int prep_total = 65536 + 65536 + 32768 + 16384 + 8192 + 4096;
    srnn_prep<<<(prep_total + 255) / 256, 256, 0, stream>>>(
        W_eh, W_hh, W_sh, W_hs, W_su, W_ha, W_hg,
        wWeh, wWhh, wWsh, wWhs, wWsu, wHead);

    // 1) time-parallel embedding projection (big WMMA GEMM)
    srnn_embproj<<<dim3(2, (T_ * B_) / MB), 256, 0, stream>>>(
        inputs, emb_W, wWeh, b_eh, mE);

    // 2) batch-parallel sequential scan, weights LDS-resident (TDM-staged)
    srnn_scan<<<NBLK, 256, 0, stream>>>(
        mE, wWhh, wWsh, wWhs, wWsu, wHead,
        b_hh, b_sh, b_hs, b_su, b_ha, b_hg, empty_elem, out);
}
